// DecodingAttentionWrapper_34127810134266
// MI455X (gfx1250) — compile-verified, matrix operand order checked by reading, not run on hardware
//
#include <hip/hip_runtime.h>
#include <hip/hip_bf16.h>
#include <math.h>

// ---------------- problem constants (from reference setup_inputs) ----------
#define B_    4
#define H_    32
#define HKV_  8
#define G_    4      // H / Hkv
#define D_    128
#define P_    128    // pages
#define S_    64     // tokens per page
#define KP_   16     // selected pages (15 top-k + last)
#define NEGF  (-1000000000.0f)
#define SM_SCALE 0.08838834764831843f   // 1/sqrt(128)

typedef _Float16 v8h  __attribute__((ext_vector_type(8)));
typedef _Float16 v16h __attribute__((ext_vector_type(16)));
typedef float    v8f  __attribute__((ext_vector_type(8)));
typedef int      vi4  __attribute__((ext_vector_type(4)));   // native vector for NT loads

// ---------------------------------------------------------------------------
// Phase 0: NeoX RoPE for q (B,H,D) and current k (B,Hkv,D).
// Writes f32 roped q, f16 roped q (WMMA A operand source), f32 roped k.
// ---------------------------------------------------------------------------
__global__ __launch_bounds__(128) void rope_kernel(
    const float* __restrict__ q, const float* __restrict__ k,
    const int* __restrict__ timestep,
    float* __restrict__ qro, _Float16* __restrict__ qhh,
    float* __restrict__ kro)
{
  const int row = blockIdx.x;          // 0..B*H-1 => q rows, then B*Hkv k rows
  const int d   = threadIdx.x;         // 0..127
  const float pos = (float)(timestep[0] - 1);
  const int i = d & 63;                // index within half
  // inv_freq = 10000^(-i/64) = exp(-i * ln(10000)/64)
  const float inv_freq = __expf(-(float)i * (9.210340371976184f / 64.0f));
  const float ang = pos * inv_freq;
  float sn, cs;
  sincosf(ang, &sn, &cs);
  const bool isQ = row < B_ * H_;
  const float* src = isQ ? (q + (long)row * D_) : (k + (long)(row - B_ * H_) * D_);
  const float x1 = src[i];
  const float x2 = src[i + 64];
  const float val = (d < 64) ? (x1 * cs - x2 * sn) : (x1 * sn + x2 * cs);
  if (isQ) {
    qro[(long)row * D_ + d] = val;
    qhh[(long)row * D_ + d] = (_Float16)val;
  } else {
    kro[(long)(row - B_ * H_) * D_ + d] = val;
  }
}

// ---------------------------------------------------------------------------
// vi4 -> v16h helper: pack 16 int32 lane-values (exact 0..126) into the
// WMMA 16-bit operand half-vector. (short) cast encourages v_cvt_f16_i16.
// ---------------------------------------------------------------------------
__device__ __forceinline__ v16h cvt16(const vi4 i0, const vi4 i1,
                                      const vi4 i2, const vi4 i3)
{
  v16h bv;
  bv[0]  = (_Float16)(short)i0.x; bv[1]  = (_Float16)(short)i0.y;
  bv[2]  = (_Float16)(short)i0.z; bv[3]  = (_Float16)(short)i0.w;
  bv[4]  = (_Float16)(short)i1.x; bv[5]  = (_Float16)(short)i1.y;
  bv[6]  = (_Float16)(short)i1.z; bv[7]  = (_Float16)(short)i1.w;
  bv[8]  = (_Float16)(short)i2.x; bv[9]  = (_Float16)(short)i2.y;
  bv[10] = (_Float16)(short)i2.z; bv[11] = (_Float16)(short)i2.w;
  bv[12] = (_Float16)(short)i3.x; bv[13] = (_Float16)(short)i3.y;
  bv[14] = (_Float16)(short)i3.z; bv[15] = (_Float16)(short)i3.w;
  return bv;
}

// ---------------------------------------------------------------------------
// Phase 1: full Q·K page scan with WMMA f32_16x16x32_f16.
// One wave per (b,hkv,page). A = roped q (rows 0..3 real, rest zero).
// Two independent token-tile accumulator chains per iteration so the
// scheduler can fill the 5-slot WMMA->WMMA RAW hazard with the other chain.
// K-cache reads are non-temporal (single-pass 134MB stream; keep L2 for the
// scores workspace that phase 3 re-reads).
// ---------------------------------------------------------------------------
__global__ __launch_bounds__(256) void qk_pagescan_kernel(
    const int* __restrict__ k_cache, const float* __restrict__ kv_scale,
    const int* __restrict__ lengths, const _Float16* __restrict__ qhh,
    float* __restrict__ scores, float* __restrict__ stats)
{
  const int bh   = blockIdx.x;              // b*HKV + h, 0..31
  const int b    = bh / HKV_;
  const int h    = bh % HKV_;
  const int wave = threadIdx.x >> 5;        // 0..7
  const int lane = threadIdx.x & 31;
  const int page = blockIdx.y * 8 + wave;   // 0..127
  const int m    = lane & 15;               // row/col within 16
  const int hiH  = lane >> 4;               // 0: K base 0, 1: K base 8
  const float scale = kv_scale[0] * SM_SCALE;   // fold dequant into score scale
  const int len = max(lengths[b], 1);

  // ---- A operand: roped q in WMMA 16-bit A layout, 4 K-chunks of 32 ----
  // lane (m, hiH): row M=m, halves K = {kb..kb+7} U {kb+16..kb+23}, kb = c*32 + 8*hiH
  v16h aop[4];
  {
    v16h z;
    #pragma unroll
    for (int i = 0; i < 16; ++i) z[i] = (_Float16)0.0f;
    const long qrow = (long)(b * H_ + h * G_ + m) * D_;
    #pragma unroll
    for (int c = 0; c < 4; ++c) {
      if (m < G_) {
        const int kb = c * 32 + hiH * 8;
        const v8h lo = *(const v8h*)(qhh + qrow + kb);
        const v8h hi = *(const v8h*)(qhh + qrow + kb + 16);
        v16h a;
        #pragma unroll
        for (int i = 0; i < 8; ++i) { a[i] = lo[i]; a[i + 8] = hi[i]; }
        aop[c] = a;
      } else {
        aop[c] = z;
      }
    }
  }

  const long pageRowBase = ((long)b * P_ + page) * S_;   // token-row index base

  float pmax[G_] = {NEGF, NEGF, NEGF, NEGF};

  #pragma unroll 1
  for (int tp = 0; tp < 2; ++tp) {            // 2 pairs of 16-token tiles
    const int t0 = tp * 2, t1 = tp * 2 + 1;
    const int* krow0 = k_cache + ((pageRowBase + t0 * 16 + m) * HKV_ + h) * (long)D_;
    const int* krow1 = k_cache + ((pageRowBase + t1 * 16 + m) * HKV_ + h) * (long)D_;
    if (tp == 0)
      __builtin_prefetch((const void*)(krow0 + 32 * HKV_ * D_), 0, 1);

    v8f acc0, acc1;
    #pragma unroll
    for (int i = 0; i < 8; ++i) { acc0[i] = 0.0f; acc1[i] = 0.0f; }

    #pragma unroll
    for (int c = 0; c < 4; ++c) {             // K chunks of 32 over D=128
      const int kb = c * 32 + hiH * 8;
      const vi4 a0 = __builtin_nontemporal_load((const vi4*)(krow0 + kb));
      const vi4 a1 = __builtin_nontemporal_load((const vi4*)(krow0 + kb + 4));
      const vi4 a2 = __builtin_nontemporal_load((const vi4*)(krow0 + kb + 16));
      const vi4 a3 = __builtin_nontemporal_load((const vi4*)(krow0 + kb + 20));
      const vi4 b0 = __builtin_nontemporal_load((const vi4*)(krow1 + kb));
      const vi4 b1 = __builtin_nontemporal_load((const vi4*)(krow1 + kb + 4));
      const vi4 b2 = __builtin_nontemporal_load((const vi4*)(krow1 + kb + 16));
      const vi4 b3 = __builtin_nontemporal_load((const vi4*)(krow1 + kb + 20));
      const v16h bv0 = cvt16(a0, a1, a2, a3);
      const v16h bv1 = cvt16(b0, b1, b2, b3);
      acc0 = __builtin_amdgcn_wmma_f32_16x16x32_f16(
          false, aop[c], false, bv0, (short)0, acc0, false, false);
      acc1 = __builtin_amdgcn_wmma_f32_16x16x32_f16(
          false, aop[c], false, bv1, (short)0, acc1, false, false);
    }

    // C layout: lanes 0-15 hold N=m, M=r in VGPR r (r=0..3 == head g)
    const int pos0 = page * S_ + t0 * 16 + m;
    const int pos1 = page * S_ + t1 * 16 + m;
    if (lane < 16) {
      #pragma unroll
      for (int g = 0; g < G_; ++g) {
        const float s0 = (pos0 < len) ? acc0[g] * scale : NEGF;
        const float s1 = (pos1 < len) ? acc1[g] * scale : NEGF;
        float* srow = scores + (((long)bh * G_ + g) * P_ + page) * S_;
        srow[t0 * 16 + m] = s0;
        srow[t1 * 16 + m] = s1;
        pmax[g] = fmaxf(pmax[g], fmaxf(s0, s1));
      }
    }
  }

  // width-16 max reduction over token lanes
  #pragma unroll
  for (int g = 0; g < G_; ++g) {
    float v = pmax[g];
    v = fmaxf(v, __shfl_xor(v, 1, 16));
    v = fmaxf(v, __shfl_xor(v, 2, 16));
    v = fmaxf(v, __shfl_xor(v, 4, 16));
    v = fmaxf(v, __shfl_xor(v, 8, 16));
    pmax[g] = v;
  }
  if (lane == 0) {
    #pragma unroll
    for (int g = 0; g < G_; ++g)
      stats[((long)bh * G_ + g) * P_ + page] = pmax[g];
  }
}

// ---------------------------------------------------------------------------
// Phase 2: per (b,hkv,g) iterative top-15 over page_stats[0..126], append 127.
// One wave per instance; descending order, lower index wins ties (top_k semantics).
// ---------------------------------------------------------------------------
__global__ __launch_bounds__(32) void topk_kernel(
    const float* __restrict__ stats, int* __restrict__ sel,
    float* __restrict__ out_idx)
{
  const int inst = blockIdx.x;          // (b*HKV+h)*G + g, 0..127
  const int lane = threadIdx.x;
  float val[4]; int idx[4];
  #pragma unroll
  for (int i = 0; i < 4; ++i) {
    const int j = lane + 32 * i;
    idx[i] = j;
    val[i] = (j < P_ - 1) ? stats[(long)inst * P_ + j] : -3.0e38f;
  }
  for (int r = 0; r < KP_ - 1; ++r) {
    float bv = val[0]; int bi = idx[0];
    #pragma unroll
    for (int i = 1; i < 4; ++i)
      if (val[i] > bv || (val[i] == bv && idx[i] < bi)) { bv = val[i]; bi = idx[i]; }
    #pragma unroll
    for (int msk = 16; msk >= 1; msk >>= 1) {
      const float ov = __shfl_xor(bv, msk, 32);
      const int   oi = __shfl_xor(bi, msk, 32);
      if (ov > bv || (ov == bv && oi < bi)) { bv = ov; bi = oi; }
    }
    if (lane == 0) {
      sel[(long)inst * KP_ + r] = bi;
      out_idx[(long)inst * KP_ + r] = (float)bi;
    }
    #pragma unroll
    for (int i = 0; i < 4; ++i) if (idx[i] == bi) val[i] = -3.0e38f;
  }
  if (lane == 0) {
    sel[(long)inst * KP_ + (KP_ - 1)] = P_ - 1;
    out_idx[(long)inst * KP_ + (KP_ - 1)] = (float)(P_ - 1);
  }
}

// ---------------------------------------------------------------------------
// Phase 3: per (b,hkv,g): gather 1024 selected scores + current-token score,
// softmax in LDS, weighted V accumulation (int32 cache, sv folded at the end).
// V-cache reads are non-temporal (single-pass stream).
// ---------------------------------------------------------------------------
__global__ __launch_bounds__(256) void attn_select_kernel(
    const float* __restrict__ scores, const int* __restrict__ sel,
    const float* __restrict__ qro, const float* __restrict__ kro,
    const float* __restrict__ v_in, const int* __restrict__ v_cache,
    const float* __restrict__ kv_scale, float* __restrict__ out)
{
  __shared__ float p_lds[KP_ * S_ + 1];   // 1025 unnormalized weights
  __shared__ float redmax[8];
  __shared__ float redsum[8];
  __shared__ float cur_acc;
  __shared__ float tot_s;
  __shared__ float mx_s;
  __shared__ float out_acc[D_];
  __shared__ int   sel_s[KP_];

  const int inst = blockIdx.x;            // (b*HKV+h)*G + g
  const int bh   = inst >> 2;
  const int g    = inst & 3;
  const int b    = bh / HKV_;
  const int h    = bh % HKV_;
  const int t    = threadIdx.x;
  const int lane = t & 31;
  const int wave = t >> 5;
  const float sv = kv_scale[1];

  if (t < KP_) sel_s[t] = sel[(long)inst * KP_ + t];
  if (t == 0) cur_acc = 0.0f;
  if (t < D_) out_acc[t] = 0.0f;
  __syncthreads();

  // current-token score partial sums (ds_add_f32)
  if (t < D_) {
    const float part = qro[(long)(b * H_ + h * G_ + g) * D_ + t] *
                       kro[(long)(b * HKV_ + h) * D_ + t];
    atomicAdd(&cur_acc, part);
  }

  // gather 4 pre-masked, pre-scaled scores per thread
  float sc[4];
  #pragma unroll
  for (int i = 0; i < 4; ++i) {
    const int j  = t + 256 * i;                 // 0..1023
    const int pg = sel_s[j >> 6];
    sc[i] = scores[((long)inst * P_ + pg) * S_ + (j & 63)];
  }
  __syncthreads();
  const float cur = cur_acc * SM_SCALE;

  // --- max reduce over 1025 values ---
  float mx = fmaxf(fmaxf(sc[0], sc[1]), fmaxf(sc[2], sc[3]));
  #pragma unroll
  for (int msk = 16; msk >= 1; msk >>= 1) mx = fmaxf(mx, __shfl_xor(mx, msk, 32));
  if (lane == 0) redmax[wave] = mx;
  __syncthreads();
  if (t == 0) {
    float mm = cur;
    for (int i = 0; i < 8; ++i) mm = fmaxf(mm, redmax[i]);
    mx_s = mm;
  }
  __syncthreads();
  mx = mx_s;

  // --- exp + sum ---
  float sum = 0.0f;
  #pragma unroll
  for (int i = 0; i < 4; ++i) {
    const float e = __expf(sc[i] - mx);
    p_lds[t + 256 * i] = e;
    sum += e;
  }
  #pragma unroll
  for (int msk = 16; msk >= 1; msk >>= 1) sum += __shfl_xor(sum, msk, 32);
  if (lane == 0) redsum[wave] = sum;
  __syncthreads();
  if (t == 0) {
    const float ec = __expf(cur - mx);
    p_lds[KP_ * S_] = ec;
    float tot = ec;
    for (int i = 0; i < 8; ++i) tot += redsum[i];
    tot_s = tot;
  }
  __syncthreads();
  const float inv = 1.0f / tot_s;

  // --- weighted V accumulation: wave owns 2 pages, lane owns 4 contiguous d ---
  float4 acc = make_float4(0.f, 0.f, 0.f, 0.f);
  #pragma unroll 1
  for (int pp = 0; pp < 2; ++pp) {
    const int slot = wave * 2 + pp;             // 0..15
    const int pg   = sel_s[slot];
    const long rowBase = ((long)b * P_ + pg) * S_;
    #pragma unroll 4
    for (int sI = 0; sI < S_; ++sI) {
      const int* vrow = v_cache + ((rowBase + sI) * HKV_ + h) * (long)D_ + lane * 4;
      const vi4 vi = __builtin_nontemporal_load((const vi4*)vrow);
      const float pw = p_lds[slot * S_ + sI];
      acc.x += pw * (float)vi.x;
      acc.y += pw * (float)vi.y;
      acc.z += pw * (float)vi.z;
      acc.w += pw * (float)vi.w;
    }
  }
  atomicAdd(&out_acc[lane * 4 + 0], acc.x * sv);
  atomicAdd(&out_acc[lane * 4 + 1], acc.y * sv);
  atomicAdd(&out_acc[lane * 4 + 2], acc.z * sv);
  atomicAdd(&out_acc[lane * 4 + 3], acc.w * sv);
  __syncthreads();

  if (t < D_) {
    const float val = out_acc[t] +
                      p_lds[KP_ * S_] * v_in[(long)(b * HKV_ + h) * D_ + t];
    out[(long)inst * D_ + t] = val * inv;
  }
}

// ---------------------------------------------------------------------------
// Host-side launcher. Inputs (setup_inputs order):
//  0 q (B,H,D) f32 | 1 k (B,Hkv,D) f32 | 2 v (B,Hkv,D) f32
//  3 kv_scale (2) f32 | 4 k_cache (B,P,S,Hkv,D) i32 | 5 v_cache i32
//  6 lengths (B) i32 | 7 timestep (1) i32
// Output: out (B,H,D) f32 ++ selected_page_idx (B,H,Kp) as f32.
// ---------------------------------------------------------------------------
extern "C" void kernel_launch(void* const* d_in, const int* in_sizes, int n_in,
                              void* d_out, int out_size, void* d_ws, size_t ws_size,
                              hipStream_t stream)
{
  (void)in_sizes; (void)n_in; (void)out_size; (void)ws_size;

  const float* q   = (const float*)d_in[0];
  const float* k   = (const float*)d_in[1];
  const float* v   = (const float*)d_in[2];
  const float* kvs = (const float*)d_in[3];
  const int*   kc  = (const int*)d_in[4];
  const int*   vc  = (const int*)d_in[5];
  const int*   len = (const int*)d_in[6];
  const int*   ts  = (const int*)d_in[7];

  // workspace layout (bytes)
  char* ws = (char*)d_ws;
  float*    qro    = (float*)(ws + 0);          //  65536 B  (B*H*D f32)
  _Float16* qhh    = (_Float16*)(ws + 65536);   //  32768 B  (B*H*D f16)
  float*    kro    = (float*)(ws + 98304);      //  16384 B  (B*Hkv*D f32)
  float*    stats  = (float*)(ws + 114688);     //  65536 B  (B*Hkv*G*P f32)
  int*      sel    = (int*)(ws + 180224);       //   8192 B  (B*Hkv*G*Kp i32)
  float*    scores = (float*)(ws + 188416);     // 4 MB     (B*Hkv*G*P*S f32)

  float* out     = (float*)d_out;
  float* out_idx = out + (long)B_ * H_ * D_;

  rope_kernel<<<dim3(B_ * H_ + B_ * HKV_), dim3(128), 0, stream>>>(
      q, k, ts, qro, qhh, kro);

  qk_pagescan_kernel<<<dim3(B_ * HKV_, P_ / 8), dim3(256), 0, stream>>>(
      kc, kvs, len, qhh, scores, stats);

  topk_kernel<<<dim3(B_ * HKV_ * G_), dim3(32), 0, stream>>>(
      stats, sel, out_idx);

  attn_select_kernel<<<dim3(B_ * HKV_ * G_), dim3(256), 0, stream>>>(
      scores, sel, qro, kro, v, vc, kvs, out);
}
